// Attention_19713899889150
// MI455X (gfx1250) — matrix operand order repaired and checked by reading, not compiled, not run on hardware
//
#include <hip/hip_runtime.h>

// ---------------------------------------------------------------------------
// Fused flash-attention for MI455X (gfx1250, wave32, WMMA).
//   out[b,q,:] = softmax(Q[b,q,:] @ K[b,:,:]^T) @ K[b,:,:]   (K == V here)
// bf16 WMMA (v_wmma_f32_16x16x32_bf16), f32 accumulate, f32 online softmax.
// MQ=32 query rows / workgroup (8 waves). Q held in registers for the whole
// kernel (8 x v16bf per wave). K/V tile staged ONCE per step into two bf16
// LDS copies: key-major kn[key][d] (S-phase B) and d-major kts[d][key]
// (O-phase B) -> each K/V byte read from global exactly once per workgroup.
// Dynamic LDS ~84 KB -> 3 workgroups per 320 KB WGP.
// ---------------------------------------------------------------------------

typedef __attribute__((ext_vector_type(16))) __bf16 v16bf;
typedef __attribute__((ext_vector_type(8)))  __bf16 v8bf;
typedef __attribute__((ext_vector_type(8)))  float  v8f;

#define BATCH 32
#define TQ    512
#define TK    2048
#define DH    512
#define MQ    32      // query rows per workgroup (2 x 16-row WMMA tiles)
#define NQT   (MQ/16) // 2 q-tiles
#define KT    32      // keys per k-step
#define NTHR  256     // 8 waves (wave32)

#define KNPAD (DH + 8)   // 520 -> 1040B row stride (16B multiple, bank-skewed)
#define KTPAD (KT + 8)   // 40  -> 80B row stride
#define SPAD  (KT + 4)   // 36 f32
#define PPAD  (KT + 8)   // 40 -> 80B

// bf16 stored as unsigned short so the host pass can sizeof() portably.
struct SmemLayout {
    unsigned short kn [KT][KNPAD];      // K/V tile, key-major bf16        (33,280 B)
    unsigned short kts[DH][KTPAD];      // K/V tile, d-major (transposed)  (40,960 B)
    float          slo[NQT][16][SPAD];  // S partial, d in [0,256)          (4,608 B)
    float          shi[NQT][16][SPAD];  // S partial, d in [256,512)        (4,608 B)
    unsigned short pbuf[NQT][16][PPAD]; // P = exp(S-m) bf16                (2,560 B)
    float          alphaBuf[NQT][16];
    float          lBuf[NQT][16];
};                                      // total ~84.3 KB (3 WGs / 320 KB WGP)

__device__ __forceinline__ __bf16 f2bf(float f) { return (__bf16)f; }

__device__ __forceinline__ unsigned pack2(float lo, float hi) {
    unsigned short a = __builtin_bit_cast(unsigned short, f2bf(lo));
    unsigned short b = __builtin_bit_cast(unsigned short, f2bf(hi));
    return (unsigned)a | ((unsigned)b << 16);
}

// A-operand (16-bit 16x32): lane-half holds k-runs {kg..kg+7} and {kg+16..kg+23}
__device__ __forceinline__ v16bf ldsA(const unsigned short* row, int off) {
    const __bf16* p = (const __bf16*)row;
    v8bf lo = *(const v8bf*)(p + off);
    v8bf hi = *(const v8bf*)(p + off + 16);
    return __builtin_shufflevector(lo, hi, 0,1,2,3,4,5,6,7,8,9,10,11,12,13,14,15);
}

// B-operand (16-bit 32x16): lane-half holds k-runs {off..off+7} and {off+16..off+23}
__device__ __forceinline__ v16bf ldsB(const unsigned short* row, int off) {
    const __bf16* p = (const __bf16*)row;
    v8bf lo = *(const v8bf*)(p + off);
    v8bf hi = *(const v8bf*)(p + off + 16);
    return __builtin_shufflevector(lo, hi, 0,1,2,3,4,5,6,7,8,9,10,11,12,13,14,15);
}

// A-operand built from a global fp32 row: halves {off..off+7},{off+16..off+23}
__device__ __forceinline__ v16bf gA(const float* p, int off) {
    float4 a = *(const float4*)(p + off);
    float4 b = *(const float4*)(p + off + 4);
    float4 c = *(const float4*)(p + off + 16);
    float4 d = *(const float4*)(p + off + 20);
    v16bf r;
    r[0]=f2bf(a.x); r[1]=f2bf(a.y); r[2]=f2bf(a.z); r[3]=f2bf(a.w);
    r[4]=f2bf(b.x); r[5]=f2bf(b.y); r[6]=f2bf(b.z); r[7]=f2bf(b.w);
    r[8]=f2bf(c.x); r[9]=f2bf(c.y); r[10]=f2bf(c.z); r[11]=f2bf(c.w);
    r[12]=f2bf(d.x); r[13]=f2bf(d.y); r[14]=f2bf(d.z); r[15]=f2bf(d.w);
    return r;
}

__global__ __launch_bounds__(NTHR)
void attn_flash_bf16(const float* __restrict__ Q,
                     const float* __restrict__ KV,
                     float* __restrict__ O)
{
    extern __shared__ __align__(16) unsigned char smem_raw[];
    SmemLayout& S = *reinterpret_cast<SmemLayout*>(smem_raw);

    const int tid = threadIdx.x;
    const int w   = tid >> 5;        // wave id 0..7
    const int qi  = w >> 2;          // q-tile 0/1
    const int wq  = w & 3;           // role within q-tile
    const int L   = tid & 31;        // lane id (wave32)
    const int hi  = L >> 4;          // lane half
    const int m   = L & 15;          // row / tile-column index
    const int kg  = hi * 8;          // A-operand k-group base

    const int b  = blockIdx.y;
    const int q0 = blockIdx.x * MQ;

    const float* kvbase = KV + (size_t)b * TK * DH;

    // S-phase split within a q-tile: wq {0,1} -> d [0,256), {2,3} -> d [256,512);
    // wq {0,2} -> keys [0,16), {1,3} -> keys [16,32) of the step tile.
    const int dlo = (wq >> 1) * 256;
    const int nb  = (wq & 1) * 16;
    float* sbuf = (wq >> 1) ? &S.shi[qi][0][0] : &S.slo[qi][0][0];

    // ---- Q slice -> registers for the whole kernel (A-operand layout) ----
    v16bf qa[8];
    {
        const float* qrow = Q + ((size_t)b * TQ + q0 + qi * 16 + m) * DH;
        #pragma unroll
        for (int c8 = 0; c8 < 8; ++c8)
            qa[c8] = gA(qrow, dlo + c8 * 32 + kg);
    }

    v8f acc[8] = {};                 // 16x128 f32 O-slice per wave (8 tiles)
    float m_run = -__builtin_inff();
    float l_run = 0.f;

    for (int k0 = 0; k0 < TK; k0 += KT) {
        // ---- stage K/V tile once: kn[key][d] and kts[d][key] (bf16) ----
        {
            const float* base = kvbase + (size_t)k0 * DH;
            const int c     = (tid & 127) * 4;        // d-column group
            const int rbase = (tid >> 7) * (KT / 4);  // 8 row-pairs per half
            #pragma unroll
            for (int i = 0; i < KT / 4; ++i) {
                int rp = rbase + i;
                float4 f0 = *(const float4*)(base + (size_t)(2*rp)   * DH + c);
                float4 f1 = *(const float4*)(base + (size_t)(2*rp+1) * DH + c);
                // key-major copy (S-phase B operands)
                unsigned a0 = pack2(f0.x, f0.y), a1 = pack2(f0.z, f0.w);
                unsigned b0 = pack2(f1.x, f1.y), b1 = pack2(f1.z, f1.w);
                *(unsigned long long*)&S.kn[2*rp  ][c] =
                    (unsigned long long)a0 | ((unsigned long long)a1 << 32);
                *(unsigned long long*)&S.kn[2*rp+1][c] =
                    (unsigned long long)b0 | ((unsigned long long)b1 << 32);
                // d-major (transposed) copy (O-phase B operands)
                *(unsigned*)&S.kts[c+0][2*rp] = pack2(f0.x, f1.x);
                *(unsigned*)&S.kts[c+1][2*rp] = pack2(f0.y, f1.y);
                *(unsigned*)&S.kts[c+2][2*rp] = pack2(f0.z, f1.z);
                *(unsigned*)&S.kts[c+3][2*rp] = pack2(f0.w, f1.w);
                // prefetch next step's tile toward the WGP (speculative)
                __builtin_prefetch(base + (size_t)(KT + 2*rp) * DH + c, 0, 3);
            }
        }
        __syncthreads();

        // ---- S partial: 16(q) x 16(keys) over this wave's 256-wide d-slice ----
        {
            const unsigned short* krow = &S.kn[nb + m][0];   // B col = key row (LDS)
            v8f s = {};
            #pragma unroll
            for (int c8 = 0; c8 < 8; ++c8) {
                v16bf bb = ldsB(krow, dlo + c8 * 32 + hi * 8);
                s = __builtin_amdgcn_wmma_f32_16x16x32_bf16(
                        false, qa[c8], false, bb, (short)0, s, false, false);
            }
            #pragma unroll
            for (int v = 0; v < 8; ++v)          // C layout: VGPR v -> row v+8*hi
                sbuf[(v + 8*hi) * SPAD + nb + m] = s[v];
        }
        __syncthreads();

        // ---- online softmax (per-lane row m of q-tile qi; redundant x4 waves) ----
        {
            float sv[KT];
            float mnew = m_run;
            #pragma unroll
            for (int j = 0; j < KT; ++j) {
                sv[j] = S.slo[qi][m][j] + S.shi[qi][m][j];
                mnew  = fmaxf(mnew, sv[j]);
            }
            float alpha = __expf(m_run - mnew);
            float lsum = 0.f;
            #pragma unroll
            for (int j = 0; j < KT; ++j) {
                sv[j] = __expf(sv[j] - mnew);    // v_exp_f32 (trans, co-executes)
                lsum += sv[j];
            }
            l_run = l_run * alpha + lsum;
            m_run = mnew;
            if (wq < 2) {                        // wq 0,1 write P cols [wq*16+kg, +8)
                int j0 = wq * 16 + kg;
                v8bf p;
                #pragma unroll
                for (int i = 0; i < 8; ++i) p[i] = f2bf(sv[j0 + i]);
                *(v8bf*)&S.pbuf[qi][m][j0] = p;
            }
            if (wq == 0 && L < 16) S.alphaBuf[qi][m] = alpha;
        }
        __syncthreads();

        // ---- rescale accumulators, O += P(16x32) x V(32x16) per d-tile ----
        {
            float a8[8];
            #pragma unroll
            for (int v = 0; v < 8; ++v) a8[v] = S.alphaBuf[qi][8*hi + v];
            v16bf pa = ldsA(&S.pbuf[qi][m][0], kg);
            #pragma unroll
            for (int t = 0; t < 8; ++t) {
                #pragma unroll
                for (int v = 0; v < 8; ++v) acc[t][v] *= a8[v];
                int dcol = wq * 128 + t * 16 + m;          // B col = d column
                v16bf vb = ldsB(&S.kts[dcol][0], hi * 8);
                acc[t] = __builtin_amdgcn_wmma_f32_16x16x32_bf16(
                             false, pa, false, vb, (short)0, acc[t], false, false);
            }
        }
        __syncthreads();   // kn/kts reused next step
    }

    if (wq == 0 && L < 16) S.lBuf[qi][m] = l_run;
    __syncthreads();

    float linv[8];
    #pragma unroll
    for (int v = 0; v < 8; ++v) linv[v] = 1.f / S.lBuf[qi][8*hi + v];

    float* orow = O + ((size_t)b * TQ + q0 + qi * 16) * DH;
    #pragma unroll
    for (int t = 0; t < 8; ++t) {
        int dcol = wq * 128 + t * 16 + m;
        #pragma unroll
        for (int v = 0; v < 8; ++v)
            orow[(size_t)(v + 8*hi) * DH + dcol] = acc[t][v] * linv[v];
    }
}

extern "C" void kernel_launch(void* const* d_in, const int* in_sizes, int n_in,
                              void* d_out, int out_size, void* d_ws, size_t ws_size,
                              hipStream_t stream) {
    (void)in_sizes; (void)n_in; (void)out_size; (void)d_ws; (void)ws_size;
    const float* Q  = (const float*)d_in[0];   // [32, 512, 512] f32
    const float* KV = (const float*)d_in[1];   // [32, 2048, 512] f32
    float* O = (float*)d_out;                  // [32, 512, 512] f32

    dim3 grid(TQ / MQ, BATCH);                 // (16, 32)
    dim3 block(NTHR);                          // 8 waves of 32
    size_t smem = sizeof(SmemLayout);          // ~84 KB dynamic LDS
    attn_flash_bf16<<<grid, block, smem, stream>>>(Q, KV, O);
}
